// GNN_89919435309131
// MI455X (gfx1250) — matrix operand order repaired and checked by reading, not compiled, not run on hardware
//
#include <hip/hip_runtime.h>
#include <hip/hip_bf16.h>
#include <stdint.h>

// ---------- types for WMMA ----------
typedef __attribute__((ext_vector_type(16))) __bf16 v16bf;
typedef __attribute__((ext_vector_type(8)))  float  v8f;

union Frag16 { v16bf v; float4 q[2]; uint32_t u[8]; };

// ---------- helpers ----------
__device__ __forceinline__ uint32_t f2bf_bits(float f) {
    uint32_t u = __builtin_bit_cast(uint32_t, f);
    // round-to-nearest-even bf16
    return (u + 0x7FFFu + ((u >> 16) & 1u)) >> 16;
}
__device__ __forceinline__ uint32_t pack_bf2(float lo, float hi) {
    return f2bf_bits(lo) | (f2bf_bits(hi) << 16);
}

// ---------- geometry ----------
#define NSAMP   128          // B*A = 16*8
#define S_IN    48
#define S1      23           // (48-3)/2+1
#define NPOS1   (S1*S1*S1)   // 12167
#define C1CH    64
#define S2      11           // (23-3)/2+1
#define NPOS2   (S2*S2*S2)   // 1331
#define C2CH    512
#define KDIM    (27*C1CH)    // 1728
#define KCHUNKS (KDIM/32)    // 54
#define PTILES2 ((NPOS2+31)/32) // 42 tiles of 32 positions

// =====================================================================
// Kernel 1: pack w2 (512,64,3,3,3) fp32 -> bf16 WMMA B-fragment layout.
// Bpack dword index t = ((kk*32+cg)*32 + lane)*8 + d
// lane<16: Klocal = 2d,2d+1 ; lane>=16: Klocal = 16+2d,17+2d ; N = cg*16+(lane&15)
// global k = kk*32+Klocal ; tap = k/64 ; ci = k%64 ; value = w2[co][ci][tap]
// =====================================================================
__global__ void k_pack_B(const float* __restrict__ w2, uint32_t* __restrict__ Bpack) {
    int t = blockIdx.x * blockDim.x + threadIdx.x;
    const int total = KCHUNKS * 32 * 32 * 8;
    if (t >= total) return;
    int d    = t & 7;
    int lane = (t >> 3) & 31;
    int cg   = (t >> 8) & 31;
    int kk   = t >> 13;
    int co = cg * 16 + (lane & 15);
    float vals[2];
    #pragma unroll
    for (int h = 0; h < 2; ++h) {
        int klocal = ((lane < 16) ? 0 : 16) + 2 * d + h;
        int k   = kk * 32 + klocal;
        int tap = k >> 6;
        int ci  = k & 63;
        vals[h] = w2[(size_t)co * (C1CH * 27) + (size_t)ci * 27 + tap];
    }
    Bpack[t] = pack_bf2(vals[0], vals[1]);
}

// =====================================================================
// Kernel 2: conv1 (1->64 ch, 3^3, stride 2) + bias + relu, write bf16
// h1 layout: [sample][z][y][x][ch]  (ch innermost, contiguous 64)
// =====================================================================
__global__ void k_conv1(const float* __restrict__ x, const float* __restrict__ w1,
                        const float* __restrict__ b1, uint32_t* __restrict__ h1u) {
    __shared__ float wl[C1CH * 27];
    for (int i = threadIdx.x; i < C1CH * 27; i += blockDim.x) wl[i] = w1[i];
    __syncthreads();

    int t = blockIdx.x * blockDim.x + threadIdx.x;
    if (t >= NSAMP * NPOS1) return;
    int sample = t / NPOS1;
    int p      = t % NPOS1;
    int oz = p / (S1 * S1);
    int oy = (p / S1) % S1;
    int ox = p % S1;

    const float* xs = x + (size_t)sample * (S_IN * S_IN * S_IN);
    float in[27];
    #pragma unroll
    for (int dz = 0; dz < 3; ++dz)
        #pragma unroll
        for (int dy = 0; dy < 3; ++dy)
            #pragma unroll
            for (int dx = 0; dx < 3; ++dx)
                in[dz * 9 + dy * 3 + dx] =
                    xs[(size_t)(2 * oz + dz) * (S_IN * S_IN) + (2 * oy + dy) * S_IN + (2 * ox + dx)];

    uint32_t* hp = h1u + (size_t)t * (C1CH / 2);
    for (int c = 0; c < C1CH; c += 2) {
        float a0 = b1[c], a1 = b1[c + 1];
        const float* w0 = &wl[c * 27];
        const float* w1r = &wl[(c + 1) * 27];
        #pragma unroll
        for (int tap = 0; tap < 27; ++tap) {
            a0 = fmaf(w0[tap], in[tap], a0);
            a1 = fmaf(w1r[tap], in[tap], a1);
        }
        a0 = a0 > 0.f ? a0 : 0.f;
        a1 = a1 > 0.f ? a1 : 0.f;
        hp[c >> 1] = pack_bf2(a0, a1);
    }
}

// =====================================================================
// Kernel 3: zero fp32 buffer
// =====================================================================
__global__ void k_zero(float* __restrict__ p, int n) {
    int t = blockIdx.x * blockDim.x + threadIdx.x;
    if (t < n) p[t] = 0.f;
}

// =====================================================================
// Kernel 4: conv2 implicit GEMM via v_wmma_f32_16x16x32_bf16,
// fused bias + relu + spatial-mean accumulation (atomicAdd into enc).
// 1 wave/block. M-tile = 32 positions (2 A frags), N = 64 channels
// (4 B frags shared by both M-subtiles) -> 8 WMMAs per K-chunk at
// ~1KB of L2 traffic per WMMA. Fully unrolled; all loads are
// base + compile-time-constant offsets; invalid rows clamped and
// masked only in the epilogue (no EXEC divergence in the hot loop).
// =====================================================================
__global__ void __launch_bounds__(32)
k_conv2_wmma(const uint16_t* __restrict__ h1, const uint32_t* __restrict__ Bpack,
             const float* __restrict__ b2, float* __restrict__ enc) {
    const int lane   = threadIdx.x;
    const int ptile  = blockIdx.x;       // 0..41 (32 positions each)
    const int sample = blockIdx.y;       // 0..127
    const int cg0    = blockIdx.z * 4;   // channel group base (16-wide groups)

    const int m = lane & 15;
    const int hi = (lane >> 4) & 1;      // which half of the 64B K-chunk

    // per-lane row base pointers for the two M-subtiles
    const char* abase[2];
    #pragma unroll
    for (int s = 0; s < 2; ++s) {
        int p = ptile * 32 + s * 16 + m;
        p = p < NPOS2 ? p : (NPOS2 - 1); // clamp: garbage rows masked in epilogue
        const int oz = p / (S2 * S2);
        const int oy = (p / S2) % S2;
        const int ox = p % S2;
        abase[s] = reinterpret_cast<const char*>(
            h1 + ((size_t)sample * NPOS1 +
                  (size_t)(2 * oz) * (S1 * S1) + (2 * oy) * S1 + (2 * ox)) * C1CH) + hi * 32;
    }
    const char* bbase = reinterpret_cast<const char*>(
        reinterpret_cast<const float4*>(Bpack) + ((size_t)cg0 * 32 + lane) * 2);

    v8f acc[2][4];
    #pragma unroll
    for (int s = 0; s < 2; ++s)
        #pragma unroll
        for (int j = 0; j < 4; ++j) acc[s][j] = (v8f){};

    #pragma unroll
    for (int kk = 0; kk < KCHUNKS; ++kk) {
        const int tap = kk >> 1;
        const int cib = (kk & 1) * 32;
        const int dz = tap / 9, dy = (tap % 9) / 3, dx = tap % 3;
        const int aoff = ((dz * (S1 * S1) + dy * S1 + dx) * C1CH + cib) * 2; // bytes

        Frag16 a[2];
        #pragma unroll
        for (int s = 0; s < 2; ++s) {
            a[s].q[0] = *reinterpret_cast<const float4*>(abase[s] + aoff);
            a[s].q[1] = *reinterpret_cast<const float4*>(abase[s] + aoff + 16);
        }

        const int boff = kk * (32 * 32 * 32);  // bytes per K-chunk of Bpack
        #pragma unroll
        for (int j = 0; j < 4; ++j) {
            Frag16 b;
            b.q[0] = *reinterpret_cast<const float4*>(bbase + boff + j * 1024);
            b.q[1] = *reinterpret_cast<const float4*>(bbase + boff + j * 1024 + 16);
            #pragma unroll
            for (int s = 0; s < 2; ++s) {
                acc[s][j] = __builtin_amdgcn_wmma_f32_16x16x32_bf16(
                    false, a[s].v, false, b.v, (short)0, acc[s][j], false, false);
            }
        }
    }

    // bias + relu + partial spatial sum, atomically folded into enc[sample][ch]
    const int mbase = (lane >> 4) << 3;  // 0 or 8
    #pragma unroll
    for (int j = 0; j < 4; ++j) {
        int co = (cg0 + j) * 16 + (lane & 15);
        float bias = b2[co];
        #pragma unroll
        for (int s = 0; s < 2; ++s) {
            float sum = 0.f;
            #pragma unroll
            for (int r = 0; r < 8; ++r) {
                int pr = ptile * 32 + s * 16 + mbase + r;
                float v = acc[s][j][r] + bias;
                v = v > 0.f ? v : 0.f;
                sum += (pr < NPOS2) ? v : 0.f;
            }
            float o = __shfl_down(sum, 16, 32);
            if (lane < 16)
                atomicAdd(&enc[(size_t)sample * C2CH + co], sum + o);
        }
    }
}

// =====================================================================
// Kernel 5: scale accumulated sums to spatial mean
// =====================================================================
__global__ void k_scale(float* __restrict__ p, int n, float s) {
    int t = blockIdx.x * blockDim.x + threadIdx.x;
    if (t < n) p[t] *= s;
}

// =====================================================================
// Kernel 6: cosine kNN (K=3) per batch of 8 agents. 1 block / batch.
// =====================================================================
__global__ void k_knn(const float* __restrict__ enc, int* __restrict__ nbr) {
    __shared__ float e[8 * C2CH];
    __shared__ float sim[64];
    int b = blockIdx.x;
    for (int i = threadIdx.x; i < 8 * C2CH; i += blockDim.x)
        e[i] = enc[(size_t)b * 8 * C2CH + i];
    __syncthreads();

    if (threadIdx.x < 64) {
        int i = threadIdx.x >> 3, j = threadIdx.x & 7;
        float d = 0.f;
        for (int c = 0; c < C2CH; ++c) d = fmaf(e[i * C2CH + c], e[j * C2CH + c], d);
        sim[i * 8 + j] = d;
    }
    __syncthreads();

    if (threadIdx.x < 8) {
        int i = threadIdx.x;
        float invi = rsqrtf(sim[i * 8 + i] + 1e-12f);
        float s[8];
        for (int j = 0; j < 8; ++j)
            s[j] = sim[i * 8 + j] * invi * rsqrtf(sim[j * 8 + j] + 1e-12f);
        bool used[8] = {false, false, false, false, false, false, false, false};
        for (int k = 0; k < 3; ++k) {
            float best = -1e30f; int bi = 0;
            for (int j = 0; j < 8; ++j)
                if (!used[j] && s[j] > best) { best = s[j]; bi = j; }
            used[bi] = true;
            nbr[(b * 8 + i) * 3 + k] = bi;
        }
    }
}

// =====================================================================
// Kernel 7: SAGE layer. 1 block per node (b,i), 256 threads.
// out[co] = agg . Wl[co] + bl[co] + h[i] . Wr[co]  (+ optional leaky relu)
// =====================================================================
__global__ void k_sage(const float* __restrict__ hin, const int* __restrict__ nbr,
                       const float* __restrict__ Wl, const float* __restrict__ bl,
                       const float* __restrict__ Wr, float* __restrict__ hout,
                       int leaky) {
    __shared__ float agg[C2CH];
    __shared__ float hrow[C2CH];
    int node = blockIdx.x;           // b*8 + i
    int n0 = nbr[node * 3 + 0], n1 = nbr[node * 3 + 1], n2 = nbr[node * 3 + 2];
    int b8 = (node >> 3) << 3;
    const float* r0 = hin + (size_t)(b8 + n0) * C2CH;
    const float* r1 = hin + (size_t)(b8 + n1) * C2CH;
    const float* r2 = hin + (size_t)(b8 + n2) * C2CH;
    const float* rs = hin + (size_t)node * C2CH;
    for (int c = threadIdx.x; c < C2CH; c += blockDim.x) {
        agg[c]  = (r0[c] + r1[c] + r2[c]) * (1.f / 3.f);
        hrow[c] = rs[c];
    }
    __syncthreads();

    for (int co = threadIdx.x; co < C2CH; co += blockDim.x) {
        const float* wl = Wl + (size_t)co * C2CH;
        const float* wr = Wr + (size_t)co * C2CH;
        float acc = bl[co];
        for (int c = 0; c < C2CH; ++c)
            acc = fmaf(agg[c], wl[c], fmaf(hrow[c], wr[c], acc));
        if (leaky) acc = acc > 0.f ? acc : 0.2f * acc;
        hout[(size_t)node * C2CH + co] = acc;
    }
}

// =====================================================================
// Kernel 8: mean-pool over agents + classifier -> (16,2)
// =====================================================================
__global__ void k_classify(const float* __restrict__ h, const float* __restrict__ Wc,
                           const float* __restrict__ bc, float* __restrict__ out) {
    int t = threadIdx.x;
    if (t >= 32) return;
    int b = t >> 1, k = t & 1;
    float acc = bc[k];
    for (int c = 0; c < C2CH; ++c) {
        float pooled = 0.f;
        #pragma unroll
        for (int i = 0; i < 8; ++i) pooled += h[(size_t)(b * 8 + i) * C2CH + c];
        acc = fmaf(pooled * 0.125f, Wc[k * C2CH + c], acc);
    }
    out[b * 2 + k] = acc;
}

// =====================================================================
// launcher
// =====================================================================
extern "C" void kernel_launch(void* const* d_in, const int* in_sizes, int n_in,
                              void* d_out, int out_size, void* d_ws, size_t ws_size,
                              hipStream_t stream) {
    const float* x   = (const float*)d_in[0];
    const float* w1  = (const float*)d_in[1];
    const float* b1  = (const float*)d_in[2];
    const float* w2  = (const float*)d_in[3];
    const float* b2  = (const float*)d_in[4];
    const float* Wl1 = (const float*)d_in[5];
    const float* bl1 = (const float*)d_in[6];
    const float* Wr1 = (const float*)d_in[7];
    const float* Wl2 = (const float*)d_in[8];
    const float* bl2 = (const float*)d_in[9];
    const float* Wr2 = (const float*)d_in[10];
    const float* Wl3 = (const float*)d_in[11];
    const float* bl3 = (const float*)d_in[12];
    const float* Wr3 = (const float*)d_in[13];
    const float* Wc  = (const float*)d_in[14];
    const float* bc  = (const float*)d_in[15];
    float* out = (float*)d_out;

    // workspace carve-up (256B aligned)
    uint8_t* ws = (uint8_t*)d_ws;
    size_t off = 0;
    auto alloc = [&](size_t bytes) { uint8_t* p = ws + off; off = (off + bytes + 255) & ~(size_t)255; return p; };
    uint16_t* h1    = (uint16_t*)alloc((size_t)NSAMP * NPOS1 * C1CH * 2);   // ~199 MB bf16
    uint32_t* Bpack = (uint32_t*)alloc((size_t)KCHUNKS * 32 * 32 * 8 * 4);  // ~1.8 MB
    float*    enc   = (float*)   alloc((size_t)NSAMP * C2CH * 4);           // 256 KB
    int*      nbr   = (int*)     alloc((size_t)NSAMP * 3 * 4);
    float*    hb0   = (float*)   alloc((size_t)NSAMP * C2CH * 4);
    float*    hb1   = (float*)   alloc((size_t)NSAMP * C2CH * 4);
    (void)ws_size; (void)in_sizes; (void)n_in; (void)out_size;

    // 1. pack conv2 weights into WMMA B-fragment layout
    {
        int total = KCHUNKS * 32 * 32 * 8;
        k_pack_B<<<(total + 255) / 256, 256, 0, stream>>>(w2, Bpack);
    }
    // 2. conv1 -> bf16 h1
    {
        int total = NSAMP * NPOS1;
        k_conv1<<<(total + 255) / 256, 256, 0, stream>>>(x, w1, b1, (uint32_t*)h1);
    }
    // 3. zero encoder accumulator
    k_zero<<<(NSAMP * C2CH + 255) / 256, 256, 0, stream>>>(enc, NSAMP * C2CH);
    // 4. conv2 WMMA + fused relu + spatial-mean accumulation
    {
        dim3 grid(PTILES2, NSAMP, C2CH / 64);
        k_conv2_wmma<<<grid, 32, 0, stream>>>(h1, Bpack, b2, enc);
    }
    // 5. sums -> means
    k_scale<<<(NSAMP * C2CH + 255) / 256, 256, 0, stream>>>(enc, NSAMP * C2CH, 1.f / (float)NPOS2);
    // 6. cosine kNN
    k_knn<<<16, 64, 0, stream>>>(enc, nbr);
    // 7-9. three SAGE layers
    k_sage<<<NSAMP, 256, 0, stream>>>(enc, nbr, Wl1, bl1, Wr1, hb0, 1);
    k_sage<<<NSAMP, 256, 0, stream>>>(hb0, nbr, Wl2, bl2, Wr2, hb1, 1);
    k_sage<<<NSAMP, 256, 0, stream>>>(hb1, nbr, Wl3, bl3, Wr3, hb0, 0);
    // 10. pool + classify
    k_classify<<<1, 32, 0, stream>>>(hb0, Wc, bc, out);
}
